// FUNASR_NANO_DECODER_MAIN_5909874999400
// MI455X (gfx1250) — matrix operand order repaired and checked by reading, not compiled, not run on hardware
//
#include <hip/hip_runtime.h>

// ---------------- model constants (from reference) ----------------
#define L_    16
#define B_    2
#define S_    1024
#define P_    1024
#define H_    16
#define KV_   2
#define G_    8      // H / KV
#define HD_   64
#define QKH_  18     // H + KV
#define NH_   20     // QKH + KV (qkv heads)
#define HID_  1024
#define DFF_  4096
#define VOC_  32000
#define T_    (P_ + S_)          // 2048 total kv length
#define QKV_OUT_ 1280            // (H + 2*KV) * HD
#define MR_   (B_ * S_)          // 2048 rows for big GEMMs

// ---------------- vector types for WMMA ----------------
typedef __attribute__((ext_vector_type(16))) __bf16    v16bf;
typedef __attribute__((ext_vector_type(8)))  __bf16    v8bf;
typedef __attribute__((ext_vector_type(16))) _Float16  v16h;
typedef __attribute__((ext_vector_type(8)))  float     v8f;
typedef __attribute__((address_space(3)))    __bf16    lds_bf16;

// float -> bf16 (round to nearest even)
__device__ __forceinline__ __bf16 f2bf(float f) {
    unsigned u = __float_as_uint(f);
    u = (u + 0x7FFFu + ((u >> 16) & 1u)) >> 16;
    unsigned short s = (unsigned short)u;
    __bf16 b;
    __builtin_memcpy(&b, &s, 2);
    return b;
}

__device__ __forceinline__ v16bf cat8(v8bf lo, v8bf hi) {
    return __builtin_shufflevector(lo, hi,
        0, 1, 2, 3, 4, 5, 6, 7, 8, 9, 10, 11, 12, 13, 14, 15);
}
__device__ __forceinline__ v16bf ld32_lds(const __bf16* p) {
    return cat8(*(const v8bf*)p, *(const v8bf*)(p + 16));
}

// issue one 16-byte async DMA global->LDS (ASYNCcnt, not LOADcnt)
__device__ __forceinline__ void async_b128(unsigned lds_byte, const __bf16* gsrc) {
    unsigned long long src = (unsigned long long)gsrc;
    asm volatile("global_load_async_to_lds_b128 %0, %1, off"
                 :: "v"(lds_byte), "v"(src) : "memory");
}
__device__ __forceinline__ void async_b128_off16(unsigned lds_byte, const __bf16* gsrc) {
    unsigned long long src = (unsigned long long)gsrc;
    asm volatile("global_load_async_to_lds_b128 %0, %1, off offset:16"
                 :: "v"(lds_byte), "v"(src) : "memory");
}
__device__ __forceinline__ void wait_async0() {
    asm volatile("s_wait_asynccnt 0x0" ::: "memory");
}

// =====================================================================
// GEMM: C[M,N] (f32) = A[M,K] (bf16 row-major) * Wbf[N,K]^T (bf16,
// row-major) + optional bias[N].
// Block = 8 waves = 128 rows x 128 cols; wave tile = 16 x 128 (8 acc).
// B slab (128 cols x 32 K, 8KB) staged in LDS via async DMA
// (global_load_async_to_lds_b128, ASYNCcnt) -- double buffered; the DMA
// for slab k+1 is in flight during the 8 wmmas of slab k and drained
// with s_wait_asynccnt right before the barrier.  A fragments are
// register-pipelined one K-step ahead.  LDS column stride padded to 40
// halfs -> bank-conflict-free ds_loads.
// =====================================================================
#define GBM 128
#define GBN 128
#define GKS 32
#define GKP 40          // padded K stride in LDS (bf16 elements)

__global__ __launch_bounds__(256)
void gemm_bf16_wmma_kernel(const __bf16* __restrict__ A, int lda,
                           const __bf16* __restrict__ Wbf, int ldw,
                           const float*  __restrict__ bias,
                           float* __restrict__ C, int ldc,
                           int M, int N, int K)
{
    __shared__ __bf16 bs[2][GBN * GKP];     // 2 x 10240 B = 20 KB

    const int lane = threadIdx.x & 31;
    const int wave = threadIdx.x >> 5;
    const int row0 = blockIdx.y * GBM + wave * 16;
    const int col0 = blockIdx.x * GBN;

    // A fragment: lane&15 = M row; lane bit4 selects K-chunk pairs.
    const int am = lane & 15;
    const int ak = (lane >> 4) << 3;    // 0 or 8
    // B fragment: lane&15 = N column; lane bit4 selects K-chunk pairs.
    const int bn = lane & 15;
    const int bk = (lane >> 4) << 3;    // 0 or 8

    int arow = row0 + am;
    if (arow >= M) arow = M - 1;        // clamp for padded lm_head tile
    const __bf16* Ap = A + (size_t)arow * lda + ak;

    // staging map: thread i -> col = i>>1 (0..127), khalf = (i&1)*16
    const int scol = threadIdx.x >> 1;
    const int skh  = (threadIdx.x & 1) << 4;
    const __bf16* Ws = Wbf + (size_t)(col0 + scol) * ldw + skh;

    // LDS byte address of this thread's staging slot (AS3 ptrtoint = offset)
    unsigned lds_byte0;
    {
        lds_bf16* p3 = (lds_bf16*)(&bs[0][0] + scol * GKP + skh);
        lds_byte0 = (unsigned)(size_t)p3;
    }
    const unsigned slab_bytes = GBN * GKP * 2;

    // ---- prologue: DMA slab 0, load first A fragment ----
    async_b128(lds_byte0, Ws);
    async_b128_off16(lds_byte0, Ws);
    v16bf a_cur = cat8(*(const v8bf*)Ap, *(const v8bf*)(Ap + 16));
    wait_async0();
    __syncthreads();

    v8f acc[8];
#pragma unroll
    for (int t = 0; t < 8; ++t) acc[t] = v8f{};

    int buf = 0;
    for (int k0 = 0; k0 < K; k0 += GKS) {
        const bool have_next = (k0 + GKS) < K;

        // kick off next slab's DMA + next A fragment before computing
        if (have_next) {
            const unsigned dst = lds_byte0 + (unsigned)(buf ^ 1) * slab_bytes;
            async_b128(dst, Ws + k0 + GKS);
            async_b128_off16(dst, Ws + k0 + GKS);
        }
        v16bf a_next;
        if (have_next) {
            a_next = cat8(*(const v8bf*)(Ap + k0 + GKS),
                          *(const v8bf*)(Ap + k0 + GKS + 16));
        }

        // 8 wmma from the LDS slab, B tiles pipelined one ahead
        const __bf16* bsl = &bs[buf][0] + bn * GKP + bk;
        v16bf bcur = ld32_lds(bsl);
#pragma unroll
        for (int t = 0; t < 8; ++t) {
            v16bf bnext;
            if (t < 7) bnext = ld32_lds(bsl + (t + 1) * 16 * GKP);
            acc[t] = __builtin_amdgcn_wmma_f32_16x16x32_bf16(
                false, a_cur, false, bcur, (short)0, acc[t], false, false);
            bcur = bnext;
        }

        if (have_next) a_cur = a_next;
        wait_async0();
        __syncthreads();
        buf ^= 1;
    }

    // ---- store C (D-layout: VGPR r -> row rb+r, lane&15 -> column) ----
    if (row0 < M) {
        const int cn = lane & 15;
        const int rb = (lane >> 4) << 3;
#pragma unroll
        for (int t = 0; t < 8; ++t) {
            int col = col0 + t * 16 + cn;
            float bv = bias ? bias[col] : 0.0f;
#pragma unroll
            for (int r = 0; r < 8; ++r) {
                C[(size_t)(row0 + rb + r) * ldc + col] = acc[t][r] + bv;
            }
        }
    }
}

// =====================================================================
// Attention: one block per (b, kv, g, 16-row q tile). Full score row
// (16 x 2048 f32 = 128 KB) lives in LDS (CDNA5 WGP has 320 KB).
//   Phase 1: S = Q(16x64,f16) x K(64x2048,f16) + mask  via wmma f16
//   Phase 2: row softmax over 2048 (cooperative, 8 threads/row)
//   Phase 3: O = P(16x2048) x V(2048x64) via wmma f16, 4-wave partials
// =====================================================================
#define ATTN_SMEM_BYTES (16 * T_ * 4 + 4 * 16 * 64 * 4 + 16 * 8 * 4)

__global__ __launch_bounds__(128)
void attention_kernel(const _Float16* __restrict__ qf,    // (B,KV,G,S,HD)
                      const _Float16* __restrict__ keys,  // (B,KV,HD,T) this layer
                      const _Float16* __restrict__ vals,  // (B,KV,T,HD) this layer
                      const float*    __restrict__ mask,  // (B,1,1,S,T)
                      float* __restrict__ out)            // (B,S,HID)
{
    extern __shared__ float smem[];
    float* sc   = smem;                     // [16][T_]
    float* pout = smem + 16 * T_;           // [4][16][64]
    float* red  = pout + 4 * 16 * 64;       // [16][8]

    const int blk = blockIdx.x;
    const int st  = blk % (S_ / 16);
    const int g   = (blk / (S_ / 16)) % G_;
    const int kv  = (blk / (S_ / 16 * G_)) % KV_;
    const int b   =  blk / (S_ / 16 * G_ * KV_);
    const int s0  = st * 16;

    const int lane = threadIdx.x & 31;
    const int wave = threadIdx.x >> 5;
    const int cn   = lane & 15;
    const int rb   = (lane >> 4) << 3;

    // ---- load Q tile fragments: a0 = K 0..31, a1 = K 32..63 ----
    const int am = lane & 15;
    const int ak = (lane >> 4) << 4;
    const _Float16* qp = qf
        + ((((size_t)b * KV_ + kv) * G_ + g) * S_ + s0 + am) * HD_ + ak;
    v16h a0, a1;
#pragma unroll
    for (int j = 0; j < 8; ++j) {
        a0[j]     = qp[j];       a0[8 + j] = qp[8 + j];
        a1[j]     = qp[32 + j];  a1[8 + j] = qp[40 + j];
    }

    const int bn = lane & 15;
    const int bk = (lane >> 4) << 4;
    const size_t kbase = ((size_t)b * KV_ + kv) * HD_ * T_;
    const size_t vbase = ((size_t)b * KV_ + kv) * (size_t)T_ * HD_;

    // ---------------- Phase 1: scores = Q x K + mask ----------------
    for (int c = wave; c < T_ / 16; c += 4) {
        const int t0 = c * 16;
        v16h b0, b1;   // K rows: b0 -> hd 0..31, b1 -> hd 32..63
#pragma unroll
        for (int j = 0; j < 16; ++j) {
            b0[j] = keys[kbase + (size_t)(bk + j) * T_ + t0 + bn];
            b1[j] = keys[kbase + (size_t)(32 + bk + j) * T_ + t0 + bn];
        }
        v8f s = v8f{};
        s = __builtin_amdgcn_wmma_f32_16x16x32_f16(false, a0, false, b0,
                                                   (short)0, s, false, false);
        s = __builtin_amdgcn_wmma_f32_16x16x32_f16(false, a1, false, b1,
                                                   (short)0, s, false, false);
#pragma unroll
        for (int r = 0; r < 8; ++r) {
            const int row = rb + r;
            const float mv = mask[((size_t)b * S_ + s0 + row) * T_ + t0 + cn];
            sc[row * T_ + t0 + cn] = s[r] + mv;
        }
    }
    __syncthreads();

    // ---------------- Phase 2: row softmax ----------------
    {
        const int row = threadIdx.x >> 3;   // 16 rows x 8 threads
        const int j   = threadIdx.x & 7;
        float mx = -3.4e38f;
        for (int i = j; i < T_; i += 8) mx = fmaxf(mx, sc[row * T_ + i]);
        red[row * 8 + j] = mx;
        __syncthreads();
        float m = -3.4e38f;
#pragma unroll
        for (int jj = 0; jj < 8; ++jj) m = fmaxf(m, red[row * 8 + jj]);
        __syncthreads();
        float ssum = 0.0f;
        for (int i = j; i < T_; i += 8) {
            float e = __expf(sc[row * T_ + i] - m);
            sc[row * T_ + i] = e;
            ssum += e;
        }
        red[row * 8 + j] = ssum;
        __syncthreads();
        float tot = 0.0f;
#pragma unroll
        for (int jj = 0; jj < 8; ++jj) tot += red[row * 8 + jj];
        const float inv = 1.0f / tot;
        for (int i = j; i < T_; i += 8) sc[row * T_ + i] *= inv;
    }
    __syncthreads();

    // ---------------- Phase 3: out = P x V ----------------
    v8f o[4];
    o[0] = v8f{}; o[1] = v8f{}; o[2] = v8f{}; o[3] = v8f{};
    for (int c = wave; c < T_ / 32; c += 4) {
        const int t0 = c * 32;
        v16h pa;     // P fragment from LDS (f32 -> f16)
#pragma unroll
        for (int j = 0; j < 8; ++j) {
            pa[j]     = (_Float16)sc[am * T_ + t0 + ak + j];
            pa[8 + j] = (_Float16)sc[am * T_ + t0 + ak + 8 + j];
        }
#pragma unroll
        for (int ht = 0; ht < 4; ++ht) {
            v16h bv;
#pragma unroll
            for (int j = 0; j < 16; ++j) {
                bv[j] = vals[vbase + (size_t)(t0 + bk + j) * HD_ + ht * 16 + bn];
            }
            o[ht] = __builtin_amdgcn_wmma_f32_16x16x32_f16(
                false, pa, false, bv, (short)0, o[ht], false, false);
        }
    }
#pragma unroll
    for (int ht = 0; ht < 4; ++ht) {
#pragma unroll
        for (int r = 0; r < 8; ++r) {
            pout[((wave * 16) + rb + r) * 64 + ht * 16 + cn] = o[ht][r];
        }
    }
    __syncthreads();

    // reduce 4 wave partials and write (B,S,HID) with head h = kv*G+g
    const int hh = kv * G_ + g;
    for (int i = threadIdx.x; i < 16 * 64; i += 128) {
        const int row = i >> 6, hd = i & 63;
        float v = pout[row * 64 + hd] + pout[(16 + row) * 64 + hd]
                + pout[(32 + row) * 64 + hd] + pout[(48 + row) * 64 + hd];
        out[((size_t)b * S_ + s0 + row) * HID_ + hh * HD_ + hd] = v;
    }
}

// =====================================================================
// Elementwise / normalization kernels
// =====================================================================
__global__ __launch_bounds__(256)
void rmsnorm_kernel(const float* __restrict__ X, __bf16* __restrict__ Y, int cols)
{
    __shared__ float sred[256];
    const int row = blockIdx.x;
    const float* x = X + (size_t)row * cols;
    float ss = 0.0f;
    for (int i = threadIdx.x; i < cols; i += 256) { float v = x[i]; ss += v * v; }
    sred[threadIdx.x] = ss;
    __syncthreads();
    for (int off = 128; off > 0; off >>= 1) {
        if (threadIdx.x < off) sred[threadIdx.x] += sred[threadIdx.x + off];
        __syncthreads();
    }
    const float inv = rsqrtf(sred[0]);   // reference: x * rsqrt(sum(x^2)), no eps
    __bf16* y = Y + (size_t)row * cols;
    for (int i = threadIdx.x; i < cols; i += 256) y[i] = f2bf(x[i] * inv);
}

// rms of last token rows, zero-padded to 16 rows for the lm_head GEMM
__global__ __launch_bounds__(256)
void rms_last_kernel(const float* __restrict__ h, __bf16* __restrict__ Y)
{
    __shared__ float sred[256];
    const int row = blockIdx.x;
    __bf16* y = Y + (size_t)row * HID_;
    if (row >= B_) {
        for (int i = threadIdx.x; i < HID_; i += 256) y[i] = f2bf(0.0f);
        return;
    }
    const float* x = h + ((size_t)row * S_ + (S_ - 1)) * HID_;
    float ss = 0.0f;
    for (int i = threadIdx.x; i < HID_; i += 256) { float v = x[i]; ss += v * v; }
    sred[threadIdx.x] = ss;
    __syncthreads();
    for (int off = 128; off > 0; off >>= 1) {
        if (threadIdx.x < off) sred[threadIdx.x] += sred[threadIdx.x + off];
        __syncthreads();
    }
    const float inv = rsqrtf(sred[0]);
    for (int i = threadIdx.x; i < HID_; i += 256) y[i] = f2bf(x[i] * inv);
}

// QK rmsnorm + rotary (rotate_half = swap halves, no negation) + scatter.
// One 64-thread block per (b, s, head).
__global__ __launch_bounds__(64)
void rope_qknorm_kernel(const float* __restrict__ qkv,   // (B,S,QKV_OUT)
                        const float* __restrict__ cosb,  // (B,S,HD)
                        const float* __restrict__ sinb,  // (B,S,HD)
                        const float* __restrict__ qkn,   // (QKH,HD) layer slice
                        _Float16* __restrict__ qf,       // (B,KV,G,S,HD)
                        _Float16* __restrict__ keys,     // (B,KV,HD,T) layer slice
                        _Float16* __restrict__ vals)     // (B,KV,T,HD) layer slice
{
    __shared__ float sh[64];
    __shared__ float sinv;
    const int blk  = blockIdx.x;
    const int head = blk % NH_;
    const int s    = (blk / NH_) % S_;
    const int b    =  blk / (NH_ * S_);
    const int hd   = threadIdx.x;

    const float v = qkv[((size_t)b * S_ + s) * QKV_OUT_ + head * HD_ + hd];

    if (head < QKH_) {
        sh[hd] = v * v;
        __syncthreads();
        for (int off = 32; off > 0; off >>= 1) {
            if (hd < off) sh[hd] += sh[hd + off];
            __syncthreads();
        }
        if (hd == 0) sinv = rsqrtf(sh[0]);
        __syncthreads();
        const float vn = v * sinv * qkn[head * HD_ + hd];
        sh[hd] = vn;
        __syncthreads();
        const float rot = sh[hd ^ 32];           // rotate_half: swap 32-halves
        const float c  = cosb[((size_t)b * S_ + s) * HD_ + hd];
        const float sn = sinb[((size_t)b * S_ + s) * HD_ + hd];
        const float o  = vn * c + rot * sn;
        if (head < H_) {
            const int kv = head / G_, g = head % G_;
            qf[((((size_t)b * KV_ + kv) * G_ + g) * S_ + s) * HD_ + hd] = (_Float16)o;
        } else {
            const int kv = head - H_;
            keys[(((size_t)b * KV_ + kv) * HD_ + hd) * T_ + (P_ + s)] = (_Float16)o;
        }
    } else {
        const int kv = head - QKH_;
        vals[((size_t)b * KV_ + kv) * (size_t)T_ * HD_ + (size_t)(P_ + s) * HD_ + hd]
            = (_Float16)v;
    }
}

__global__ void copy_past_keys_kernel(const _Float16* __restrict__ pk,  // (B,KV,HD,P)
                                      _Float16* __restrict__ keys)      // (B,KV,HD,T)
{
    const int i = blockIdx.x * blockDim.x + threadIdx.x;
    const int total = B_ * KV_ * HD_ * P_;
    if (i >= total) return;
    const int p = i % P_;
    const int r = i / P_;
    keys[(size_t)r * T_ + p] = pk[i];
}

__global__ void copy_past_vals_kernel(const _Float16* __restrict__ pv,  // (B,KV,P,HD)
                                      _Float16* __restrict__ vals)      // (B,KV,T,HD)
{
    const int i = blockIdx.x * blockDim.x + threadIdx.x;
    const int total = B_ * KV_ * P_ * HD_;
    if (i >= total) return;
    const int c = i % (P_ * HD_);
    const int r = i / (P_ * HD_);
    vals[(size_t)r * (size_t)T_ * HD_ + c] = pv[i];
}

__global__ void copy_f32_kernel(const float* __restrict__ src, float* __restrict__ dst, int n)
{
    const int i = blockIdx.x * blockDim.x + threadIdx.x;
    if (i < n) dst[i] = src[i];
}

__global__ void add_inplace_kernel(float* __restrict__ h, const float* __restrict__ d, int n)
{
    const int i = blockIdx.x * blockDim.x + threadIdx.x;
    if (i < n) h[i] += d[i];
}

// streaming f32 -> bf16 (used for activations AND per-layer weight staging)
__global__ void f32_to_bf16_kernel(const float* __restrict__ src, __bf16* __restrict__ dst, int n)
{
    const int i = blockIdx.x * blockDim.x + threadIdx.x;
    if (i < n) dst[i] = f2bf(src[i]);
}

// act = gelu_tanh(gate) * up  ->  bf16  (gu is [rows, 2*DFF])
__global__ void gelu_up_kernel(const float* __restrict__ gu, __bf16* __restrict__ act, int n)
{
    const int i = blockIdx.x * blockDim.x + threadIdx.x;
    if (i >= n) return;
    const int row = i / DFF_;
    const int col = i % DFF_;
    const float g = gu[(size_t)row * (2 * DFF_) + col];
    const float u = gu[(size_t)row * (2 * DFF_) + DFF_ + col];
    const float t = tanhf(0.7978845608028654f * (g + 0.044715f * g * g * g));
    act[i] = f2bf(0.5f * g * (1.0f + t) * u);
}

__global__ void copy_logits_kernel(const float* __restrict__ lm_out, float* __restrict__ logits)
{
    const int i = blockIdx.x * blockDim.x + threadIdx.x;
    if (i < B_ * VOC_) logits[i] = lm_out[(size_t)(i / VOC_) * VOC_ + (i % VOC_)];
}

// =====================================================================
// Orchestration
// =====================================================================
static inline int ceil_div(int a, int b) { return (a + b - 1) / b; }

extern "C" void kernel_launch(void* const* d_in, const int* in_sizes, int n_in,
                              void* d_out, int out_size, void* d_ws, size_t ws_size,
                              hipStream_t stream)
{
    (void)in_sizes; (void)n_in; (void)out_size; (void)ws_size;

    const _Float16* past_keys   = (const _Float16*)d_in[0];
    const _Float16* past_values = (const _Float16*)d_in[1];
    const float* hidden    = (const float*)d_in[2];
    const float* cosb      = (const float*)d_in[3];
    const float* sinb      = (const float*)d_in[4];
    const float* mask      = (const float*)d_in[5];
    const float* qkv_w     = (const float*)d_in[6];
    const float* qkv_b     = (const float*)d_in[7];
    const float* qk_norm_w = (const float*)d_in[8];
    const float* o_w       = (const float*)d_in[9];
    const float* gate_up_w = (const float*)d_in[10];
    const float* down_w    = (const float*)d_in[11];
    const float* lm_head_w = (const float*)d_in[12];

    // outputs: [keys f16][values f16][logits f32]
    _Float16* keys_out = (_Float16*)d_out;
    _Float16* vals_out = keys_out + (size_t)L_ * B_ * KV_ * HD_ * T_;
    float*    logits   = (float*)(vals_out + (size_t)L_ * B_ * KV_ * T_ * HD_);

    // workspace carve-out
    char* ws = (char*)d_ws;
    size_t off = 0;
    auto carve = [&](size_t bytes) -> void* {
        void* p = ws + off;
        off += (bytes + 255) & ~(size_t)255;
        return p;
    };
    float*    h        = (float*)   carve((size_t)MR_ * HID_ * 4);
    __bf16*   xnorm    = (__bf16*)  carve((size_t)MR_ * HID_ * 2);
    float*    qkvb     = (float*)   carve((size_t)MR_ * QKV_OUT_ * 4);
    _Float16* qf       = (_Float16*)carve((size_t)B_ * H_ * S_ * HD_ * 2);
    float*    attn_out = (float*)   carve((size_t)MR_ * HID_ * 4);
    float*    proj     = (float*)   carve((size_t)MR_ * HID_ * 4);
    float*    gu       = (float*)   carve((size_t)MR_ * 2 * DFF_ * 4);
    __bf16*   act      = (__bf16*)  carve((size_t)MR_ * DFF_ * 2);
    __bf16*   lm_in    = (__bf16*)  carve((size_t)16 * HID_ * 2);
    float*    lm_out   = (float*)   carve((size_t)16 * VOC_ * 4);
    // bf16 weight staging buffer: sized for the largest weight (lm_head)
    __bf16*   wstage   = (__bf16*)  carve((size_t)VOC_ * HID_ * 2);

    // allow 148KB dynamic LDS for the attention kernel
    hipFuncSetAttribute((const void*)attention_kernel,
                        hipFuncAttributeMaxDynamicSharedMemorySize,
                        ATTN_SMEM_BYTES);

    const int EW = 256;
    const int nHID = MR_ * HID_;

    copy_f32_kernel<<<ceil_div(nHID, EW), EW, 0, stream>>>(hidden, h, nHID);

    for (int l = 0; l < L_; ++l) {
        const _Float16* pkL = past_keys   + (size_t)l * B_ * KV_ * HD_ * P_;
        const _Float16* pvL = past_values + (size_t)l * B_ * KV_ * P_ * HD_;
        _Float16* keysL = keys_out + (size_t)l * B_ * KV_ * HD_ * T_;
        _Float16* valsL = vals_out + (size_t)l * B_ * KV_ * T_ * HD_;

        // x = rms(h)
        rmsnorm_kernel<<<MR_, 256, 0, stream>>>(h, xnorm, HID_);

        // qkv = x @ qkv_w^T + qkv_b   (stage weights to bf16 first)
        {
            const int nw = QKV_OUT_ * HID_;
            f32_to_bf16_kernel<<<ceil_div(nw, EW), EW, 0, stream>>>(
                qkv_w + (size_t)l * nw, wstage, nw);
            gemm_bf16_wmma_kernel<<<dim3(QKV_OUT_ / GBN, ceil_div(MR_, GBM)), 256, 0, stream>>>(
                xnorm, HID_, wstage, HID_,
                qkv_b + (size_t)l * QKV_OUT_, qkvb, QKV_OUT_, MR_, QKV_OUT_, HID_);
        }

        // KV cache: past copy + new append
        copy_past_keys_kernel<<<ceil_div(B_ * KV_ * HD_ * P_, EW), EW, 0, stream>>>(pkL, keysL);
        copy_past_vals_kernel<<<ceil_div(B_ * KV_ * P_ * HD_, EW), EW, 0, stream>>>(pvL, valsL);
        rope_qknorm_kernel<<<B_ * S_ * NH_, 64, 0, stream>>>(
            qkvb, cosb, sinb, qk_norm_w + (size_t)l * QKH_ * HD_, qf, keysL, valsL);

        // attention
        attention_kernel<<<B_ * KV_ * G_ * (S_ / 16), 128, ATTN_SMEM_BYTES, stream>>>(
            qf, keysL, valsL, mask, attn_out);

        // o projection + residual
        f32_to_bf16_kernel<<<ceil_div(nHID, EW), EW, 0, stream>>>(attn_out, xnorm, nHID);
        {
            const int nw = HID_ * HID_;
            f32_to_bf16_kernel<<<ceil_div(nw, EW), EW, 0, stream>>>(
                o_w + (size_t)l * nw, wstage, nw);
            gemm_bf16_wmma_kernel<<<dim3(HID_ / GBN, ceil_div(MR_, GBM)), 256, 0, stream>>>(
                xnorm, HID_, wstage, HID_, nullptr, proj, HID_, MR_, HID_, HID_);
        }
        add_inplace_kernel<<<ceil_div(nHID, EW), EW, 0, stream>>>(h, proj, nHID);

        // MLP
        rmsnorm_kernel<<<MR_, 256, 0, stream>>>(h, xnorm, HID_);
        {
            const int nw = 2 * DFF_ * HID_;
            f32_to_bf16_kernel<<<ceil_div(nw, EW), EW, 0, stream>>>(
                gate_up_w + (size_t)l * nw, wstage, nw);
            gemm_bf16_wmma_kernel<<<dim3(2 * DFF_ / GBN, ceil_div(MR_, GBM)), 256, 0, stream>>>(
                xnorm, HID_, wstage, HID_, nullptr, gu, 2 * DFF_, MR_, 2 * DFF_, HID_);
        }
        gelu_up_kernel<<<ceil_div(MR_ * DFF_, EW), EW, 0, stream>>>(gu, act, MR_ * DFF_);
        {
            const int nw = HID_ * DFF_;
            f32_to_bf16_kernel<<<ceil_div(nw, EW), EW, 0, stream>>>(
                down_w + (size_t)l * nw, wstage, nw);
            gemm_bf16_wmma_kernel<<<dim3(HID_ / GBN, ceil_div(MR_, GBM)), 256, 0, stream>>>(
                act, DFF_, wstage, DFF_, nullptr, proj, HID_, MR_, HID_, DFF_);
        }
        add_inplace_kernel<<<ceil_div(nHID, EW), EW, 0, stream>>>(h, proj, nHID);
    }

    // lm head on last token (M padded 2 -> 16)
    rms_last_kernel<<<16, 256, 0, stream>>>(h, lm_in);
    {
        const int nw = VOC_ * HID_;
        f32_to_bf16_kernel<<<ceil_div(nw, EW), EW, 0, stream>>>(lm_head_w, wstage, nw);
        gemm_bf16_wmma_kernel<<<dim3(VOC_ / GBN, 1), 256, 0, stream>>>(
            lm_in, HID_, wstage, HID_, nullptr, lm_out, VOC_, 16, VOC_, HID_);
    }
    copy_logits_kernel<<<ceil_div(B_ * VOC_, EW), EW, 0, stream>>>(lm_out, logits);
}
